// StructureEncoder_66700842107560
// MI455X (gfx1250) — compile-verified, matrix operand order checked by reading, not run on hardware
//
#include <hip/hip_runtime.h>
#include <hip/hip_bf16.h>

typedef __attribute__((ext_vector_type(16))) _Float16 v16h;
typedef __attribute__((ext_vector_type(8)))  float    v8f;

#define HID      512
#define LAT      256
#define HEADS    4
#define HD       128            // head dim = HID/HEADS
#define N_ATOMS  4096
#define N_MOTIFS 1024
#define E_ATOM   32768
#define E_MOTIF  8192

// ---------------------------------------------------------------------------
// CDNA5 async global->LDS copy (ASYNCcnt) via inline asm.
// LDS operand: generic pointer low 32 bits == LDS byte offset (flat LDS
// aperture truncates addr[31:0], CDNA5 ISA).
// ---------------------------------------------------------------------------
__device__ __forceinline__ void async_copy_b128(const void* gptr, void* lptr) {
  unsigned lds_off = (unsigned)(unsigned long long)lptr;
  asm volatile("global_load_async_to_lds_b128 %0, %1, off"
               :: "v"(lds_off), "v"(gptr) : "memory");
}
__device__ __forceinline__ void wait_async_zero() {
  asm volatile("s_wait_asynccnt 0" ::: "memory");
}

// ---------------------------------------------------------------------------
// Fragment loads.  16-bit A-matrix 16x32 layout (CDNA5 ISA):
//   lane l: row = l&15, khalf = l>>4
//   VGPR0-3 : K = khalf*8 + {0..7}      VGPR4-7 : K = 16 + khalf*8 + {0..7}
// (B operand of X@W^T with row-major W[N,K] loads with the same pattern.)
// ---------------------------------------------------------------------------
__device__ __forceinline__ v16h ld_frag_f32(const float* __restrict__ p,
                                            long rowbase, int ld, int k0, int lane) {
  v16h f;
  long r  = rowbase + (lane & 15);
  int  kh = lane >> 4;
  const float* s0 = p + r * (long)ld + k0 + kh * 8;
  #pragma unroll
  for (int i = 0; i < 8; ++i) { f[i] = (_Float16)s0[i]; f[8+i] = (_Float16)s0[16+i]; }
  return f;
}

__device__ __forceinline__ v16h ld_frag_f32_guard(const float* __restrict__ p,
                                                  long rowbase, int nrows, int ld,
                                                  int k0, int lane) {
  v16h f;
  long r  = rowbase + (lane & 15);
  int  kh = lane >> 4;
  if (r < nrows) {
    const float* s0 = p + r * (long)ld + k0 + kh * 8;
    #pragma unroll
    for (int i = 0; i < 8; ++i) { f[i] = (_Float16)s0[i]; f[8+i] = (_Float16)s0[16+i]; }
  } else {
    #pragma unroll
    for (int i = 0; i < 16; ++i) f[i] = (_Float16)0.f;
  }
  return f;
}

// ---------------------------------------------------------------------------
// Main GEMM: Out[M,N] = act(X[M,K] @ Wh[N,K]^T + bias), Wh pre-converted f16.
// M%64==0, N%64==0, K%32==0.  block = 128 threads (4 waves); block tile 64x64;
// wave tile 16x64.  The f16 W k-slab [64 x 32] is async-copied into a
// double-buffered LDS tile (next slab staged while current one is consumed);
// one A fragment feeds 4 WMMAs.  grid = (N/64, M/64).
// ---------------------------------------------------------------------------
#define WT_LDH 40   // padded LDS row stride in halves: 80B, 16B-aligned
__device__ __forceinline__ void stage_w(const _Float16* __restrict__ W,
                                        _Float16* dst, int nbase0, int K, int k0,
                                        int tid) {
  // 64 rows x 32 halves = 256 chunks of 8 halves (16B); 2 per thread
  #pragma unroll
  for (int j = 0; j < 2; ++j) {
    int ci  = tid + 128 * j;
    int row = ci >> 2;
    int c8  = (ci & 3) * 8;
    async_copy_b128(W + (long)(nbase0 + row) * K + k0 + c8,
                    dst + row * WT_LDH + c8);
  }
}

__global__ void k_linear_big(const float* __restrict__ X, const _Float16* __restrict__ W,
                             const float* __restrict__ bias, float* __restrict__ Out,
                             int M, int N, int K, int act) {
  __shared__ _Float16 wtile[2][64 * WT_LDH];
  int tid  = threadIdx.x;
  int wave = tid >> 5;
  int lane = tid & 31;
  long mbase  = ((long)blockIdx.y * 4 + wave) * 16;
  int  nbase0 = blockIdx.x * 64;

  v8f acc[4];
  #pragma unroll
  for (int t = 0; t < 4; ++t) acc[t] = (v8f){};

  int r  = lane & 15;
  int kh = lane >> 4;

  stage_w(W, wtile[0], nbase0, K, 0, tid);           // prologue stage

  for (int k0 = 0; k0 < K; k0 += 32) {
    int cur = (k0 >> 5) & 1;
    wait_async_zero();                               // own copies done
    __syncthreads();                                 // everyone's copies done,
                                                     // other buffer free to write
    if (k0 + 32 < K)
      stage_w(W, wtile[cur ^ 1], nbase0, K, k0 + 32, tid);

    v16h a = ld_frag_f32(X, mbase, K, k0, lane);

    #pragma unroll
    for (int t = 0; t < 4; ++t) {
      v16h b;
      const _Float16* s0 = &wtile[cur][(t * 16 + r) * WT_LDH + kh * 8];
      #pragma unroll
      for (int i = 0; i < 8; ++i) { b[i] = s0[i]; b[8 + i] = s0[16 + i]; }
      acc[t] = __builtin_amdgcn_wmma_f32_16x16x32_f16(false, a, false, b,
                                                      (short)0, acc[t], false, false);
    }
  }

  // C layout: lane l holds col n = ntile + (l&15); VGPR rr holds row mbase+rr+8*(l>>4)
  int mh = kh * 8;
  #pragma unroll
  for (int t = 0; t < 4; ++t) {
    int   n  = nbase0 + t * 16 + r;
    float bn = bias ? bias[n] : 0.f;
    #pragma unroll
    for (int rr = 0; rr < 8; ++rr) {
      float v = acc[t][rr] + bn;
      if (act) v = fmaxf(v, 0.f);
      Out[(mbase + mh + rr) * (long)N + n] = v;
    }
  }
}

// ---------------------------------------------------------------------------
// Small GEMM (only the M=1 projection head): guarded rows, f32 W, no LDS.
// grid = (N/16, ceil(M/64)), block = 128.
// ---------------------------------------------------------------------------
__global__ void k_linear_small(const float* __restrict__ X, const float* __restrict__ W,
                               const float* __restrict__ bias, float* __restrict__ Out,
                               int M, int N, int K, int act) {
  int wave = threadIdx.x >> 5;
  int lane = threadIdx.x & 31;
  long mbase = ((long)blockIdx.y * 4 + wave) * 16;
  int  nbase = blockIdx.x * 16;
  if (mbase >= M) return;

  v8f acc = {};
  for (int k0 = 0; k0 < K; k0 += 32) {
    v16h a = ld_frag_f32_guard(X, mbase, M, K, k0, lane);
    v16h b = ld_frag_f32_guard(W, nbase, N, K, k0, lane);
    acc = __builtin_amdgcn_wmma_f32_16x16x32_f16(false, a, false, b,
                                                 (short)0, acc, false, false);
  }
  int   n  = nbase + (lane & 15);
  float bn = bias ? bias[n] : 0.f;
  int   mh = (lane >> 4) * 8;
  #pragma unroll
  for (int rr = 0; rr < 8; ++rr) {
    long m = mbase + mh + rr;
    if (m < M) {
      float v = acc[rr] + bn;
      if (act) v = fmaxf(v, 0.f);
      Out[m * (long)N + n] = v;
    }
  }
}

// ---------------------------------------------------------------------------
// Elementwise / scatter helpers
// ---------------------------------------------------------------------------
__global__ void k_fill(float* __restrict__ p, float v, long n) {
  long i = (long)blockIdx.x * blockDim.x + threadIdx.x;
  if (i < n) p[i] = v;
}

__global__ void k_add(const float* __restrict__ a, const float* __restrict__ b,
                      float* __restrict__ o, long n) {
  long i = (long)blockIdx.x * blockDim.x + threadIdx.x;
  if (i < n) o[i] = a[i] + b[i];
}

__global__ void k_deg(const int* __restrict__ dst, float* __restrict__ deg, int E) {
  int e = blockIdx.x * blockDim.x + threadIdx.x;
  if (e < E) atomicAdd(&deg[dst[e]], 1.0f);
}

__global__ void k_rsqrt(float* __restrict__ d, int n) {
  int i = blockIdx.x * blockDim.x + threadIdx.x;
  if (i < n) d[i] = rsqrtf(d[i]);          // deg >= 1 (self-loop)
}

__global__ void k_gcn_scatter(const float* __restrict__ xw, const int* __restrict__ src,
                              const int* __restrict__ dst, const float* __restrict__ dinv,
                              float* __restrict__ out, int E, int ncol) {
  long t = (long)blockIdx.x * blockDim.x + threadIdx.x;
  int chunks = ncol >> 2;
  long e = t / chunks;
  int  c = (int)(t % chunks) * 4;
  if (e < E) {
    int s = src[e], d = dst[e];
    float nrm = dinv[s] * dinv[d];
    #pragma unroll
    for (int i = 0; i < 4; ++i)
      atomicAdd(&out[(long)d * ncol + c + i], xw[(long)s * ncol + c + i] * nrm);
  }
}

__global__ void k_gcn_finish(const float* __restrict__ accum, const float* __restrict__ xw,
                             const float* __restrict__ dinv, const float* __restrict__ b,
                             float* __restrict__ out, int n, int ncol) {
  long t = (long)blockIdx.x * blockDim.x + threadIdx.x;
  if (t < (long)n * ncol) {
    int i = (int)(t / ncol), c = (int)(t % ncol);
    float di = dinv[i];
    float v = accum[t] + xw[t] * di * di + b[c];
    out[t] = fmaxf(v, 0.f);
  }
}

__global__ void k_gin_scatter(const float* __restrict__ x, const int* __restrict__ src,
                              const int* __restrict__ dst, float* __restrict__ agg,
                              int E, int ncol) {
  long t = (long)blockIdx.x * blockDim.x + threadIdx.x;
  int chunks = ncol >> 2;
  long e = t / chunks;
  int  c = (int)(t % chunks) * 4;
  if (e < E) {
    int s = src[e], d = dst[e];
    #pragma unroll
    for (int i = 0; i < 4; ++i)
      atomicAdd(&agg[(long)d * ncol + c + i], x[(long)s * ncol + c + i]);
  }
}

__global__ void k_f32_to_f16(const float* __restrict__ x, _Float16* __restrict__ y, long n) {
  long i = (long)blockIdx.x * blockDim.x + threadIdx.x;
  if (i < n) y[i] = (_Float16)x[i];
}

__global__ void k_transpose_f16(const float* __restrict__ x, _Float16* __restrict__ y,
                                int L, int NC) {
  long i = (long)blockIdx.x * blockDim.x + threadIdx.x;
  if (i < (long)L * NC) {
    int row = (int)(i / NC), col = (int)(i % NC);
    y[(long)col * L + row] = (_Float16)x[i];
  }
}

__global__ void k_colmean(const float* __restrict__ x, float* __restrict__ out,
                          int L, int NC) {
  int c = blockIdx.x * blockDim.x + threadIdx.x;
  if (c < NC) {
    float s = 0.f;
    for (int i = 0; i < L; ++i) s += x[(long)i * NC + c];
    out[c] = s / (float)L;
  }
}

// ---------------------------------------------------------------------------
// Flash attention: 1 wave per (16-query tile, head), KV streamed in 32-row
// chunks.  S = Q K^T via chained WMMAs; online softmax with half-wave shuffle
// reductions; P relaid C->A layout through LDS; O += P V from f16 V^T.
// ---------------------------------------------------------------------------
__global__ void k_flash(const _Float16* __restrict__ Qh, const _Float16* __restrict__ Kh,
                        const _Float16* __restrict__ VT, float* __restrict__ Out, int L) {
  __shared__ _Float16 pbuf[16][40];
  const float scale = 0.08838834764831845f;   // 1/sqrt(128)
  int lane = threadIdx.x;
  int qt = blockIdx.x, h = blockIdx.y;
  int r  = lane & 15;
  int kh = lane >> 4;

  v16h qf[4];
  {
    const _Float16* qp = Qh + ((long)qt * 16 + r) * HID + h * HD;
    #pragma unroll
    for (int ks = 0; ks < 4; ++ks) {
      const _Float16* s0 = qp + ks * 32 + kh * 8;
      #pragma unroll
      for (int i = 0; i < 8; ++i) { qf[ks][i] = s0[i]; qf[ks][8 + i] = s0[16 + i]; }
    }
  }

  v8f o[8];
  #pragma unroll
  for (int t = 0; t < 8; ++t) o[t] = (v8f){};
  float mrow[8], lrow[8];
  #pragma unroll
  for (int i = 0; i < 8; ++i) { mrow[i] = -1e30f; lrow[i] = 0.f; }

  for (int j = 0; j < L; j += 32) {
    v8f s[2];
    #pragma unroll
    for (int t = 0; t < 2; ++t) {
      v8f acc = {};
      const _Float16* kp = Kh + ((long)(j + t * 16 + r)) * HID + h * HD;
      #pragma unroll
      for (int ks = 0; ks < 4; ++ks) {
        v16h bf;
        const _Float16* s0 = kp + ks * 32 + kh * 8;
        #pragma unroll
        for (int i = 0; i < 8; ++i) { bf[i] = s0[i]; bf[8 + i] = s0[16 + i]; }
        acc = __builtin_amdgcn_wmma_f32_16x16x32_f16(false, qf[ks], false, bf,
                                                     (short)0, acc, false, false);
      }
      s[t] = acc;
    }

    #pragma unroll
    for (int rr = 0; rr < 8; ++rr) {
      float v0 = s[0][rr] * scale, v1 = s[1][rr] * scale;
      float mx = fmaxf(v0, v1);
      #pragma unroll
      for (int d = 1; d < 16; d <<= 1) mx = fmaxf(mx, __shfl_xor(mx, d, 32));
      float nm   = fmaxf(mrow[rr], mx);
      float corr = __expf(mrow[rr] - nm);
      float p0   = __expf(v0 - nm);
      float p1   = __expf(v1 - nm);
      s[0][rr] = p0; s[1][rr] = p1;
      float rs = p0 + p1;
      #pragma unroll
      for (int d = 1; d < 16; d <<= 1) rs += __shfl_xor(rs, d, 32);
      lrow[rr] = lrow[rr] * corr + rs;
      mrow[rr] = nm;
      #pragma unroll
      for (int t = 0; t < 8; ++t) o[t][rr] *= corr;
    }

    #pragma unroll
    for (int rr = 0; rr < 8; ++rr) {
      pbuf[rr + 8 * kh][r]      = (_Float16)s[0][rr];
      pbuf[rr + 8 * kh][16 + r] = (_Float16)s[1][rr];
    }
    __syncthreads();
    v16h pf;
    #pragma unroll
    for (int i = 0; i < 8; ++i) {
      pf[i]     = pbuf[r][kh * 8 + i];
      pf[8 + i] = pbuf[r][16 + kh * 8 + i];
    }
    __syncthreads();

    #pragma unroll
    for (int t = 0; t < 8; ++t) {
      v16h vf;
      const _Float16* vp = VT + ((long)(h * HD + t * 16 + r)) * L + j + kh * 8;
      #pragma unroll
      for (int i = 0; i < 8; ++i) { vf[i] = vp[i]; vf[8 + i] = vp[16 + i]; }
      o[t] = __builtin_amdgcn_wmma_f32_16x16x32_f16(false, pf, false, vf,
                                                    (short)0, o[t], false, false);
    }
  }

  #pragma unroll
  for (int rr = 0; rr < 8; ++rr) {
    long m = (long)qt * 16 + rr + 8 * kh;
    float inv = 1.f / lrow[rr];
    #pragma unroll
    for (int t = 0; t < 8; ++t)
      Out[m * HID + h * HD + t * 16 + r] = o[t][rr] * inv;
  }
}

// ---------------------------------------------------------------------------
// Host-side orchestration
// ---------------------------------------------------------------------------
static inline int cdiv(long a, long b) { return (int)((a + b - 1) / b); }

static void linear_h(const float* X, const _Float16* Wh, const float* B, float* Out,
                     int M, int N, int K, int act, hipStream_t s) {
  dim3 g(N / 64, M / 64);
  k_linear_big<<<g, 128, 0, s>>>(X, Wh, B, Out, M, N, K, act);
}
static void zero(float* p, long n, hipStream_t s) {
  k_fill<<<cdiv(n, 256), 256, 0, s>>>(p, 0.f, n);
}
static void cvt16(const float* x, _Float16* y, long n, hipStream_t s) {
  k_f32_to_f16<<<cdiv(n, 256), 256, 0, s>>>(x, y, n);
}

extern "C" void kernel_launch(void* const* d_in, const int* in_sizes, int n_in,
                              void* d_out, int out_size, void* d_ws, size_t ws_size,
                              hipStream_t stream) {
  const float* atom_f  = (const float*)d_in[0];
  /* d_in[1] bond_features: dead (bond_enc result unused by the reference) */
  const float* motif_f = (const float*)d_in[2];
  const int*   ei      = (const int*)d_in[3];
  const int*   mei     = (const int*)d_in[4];
  const float* ae_w = (const float*)d_in[5],  * ae_b = (const float*)d_in[6];
  /* d_in[7..8] bond_enc w/b: dead */
  const float* me_w = (const float*)d_in[9],  * me_b = (const float*)d_in[10];
  const float* gcn_w[3], * gcn_b[3];
  for (int i = 0; i < 3; ++i) { gcn_w[i] = (const float*)d_in[11 + 2*i]; gcn_b[i] = (const float*)d_in[12 + 2*i]; }
  const float* gin_w1[3], * gin_b1[3], * gin_w2[3], * gin_b2[3];
  for (int i = 0; i < 3; ++i) {
    gin_w1[i] = (const float*)d_in[17 + 4*i]; gin_b1[i] = (const float*)d_in[18 + 4*i];
    gin_w2[i] = (const float*)d_in[19 + 4*i]; gin_b2[i] = (const float*)d_in[20 + 4*i];
  }
  const float* qw = (const float*)d_in[29], * qb = (const float*)d_in[30];
  const float* kw = (const float*)d_in[31], * kb = (const float*)d_in[32];
  const float* vw = (const float*)d_in[33], * vb = (const float*)d_in[34];
  const float* ow = (const float*)d_in[35], * ob = (const float*)d_in[36];
  const float* pw1 = (const float*)d_in[37], * pb1 = (const float*)d_in[38];
  const float* pw2 = (const float*)d_in[39], * pb2 = (const float*)d_in[40];

  const int* src  = ei,  * dst  = ei + E_ATOM;
  const int* msrc = mei, * mdst = mei + E_MOTIF;

  // --- workspace carve (floats) ---
  const long A = (long)N_ATOMS * HID;
  float* ws   = (float*)d_ws;
  float* x    = ws;                             // atom hidden
  float* t0   = x   + A;                        // xw / q / attn core
  float* t1   = t0  + A;                        // agg / k / attn out
  float* t2   = t1  + A;                        // v
  float* m    = t2  + A;                        // motif hidden  [1024*512]
  float* mt0  = m   + (long)N_MOTIFS * HID;     // gin hidden    [1024*1024]
  float* mt1  = mt0 + (long)N_MOTIFS * 2 * HID; // gin agg       [1024*512]
  float* mt2  = mt1 + (long)N_MOTIFS * HID;     // gin x+agg     [1024*512]
  float* deg  = mt2 + (long)N_MOTIFS * HID;     // [4096]
  float* comb = deg + N_ATOMS;                  // [1024]
  float* hprj = comb + 2 * HID;                 // [512]
  _Float16* qh = (_Float16*)(hprj + HID);       // f16 staging, A halves each
  _Float16* kh = qh + A;
  _Float16* vT = kh + A;
  // f16 weights (converted once, reused by every GEMM tile)
  _Float16* p16 = vT + A;
  _Float16* ae_wh = p16;                 p16 += (long)HID * 128;
  _Float16* me_wh = p16;                 p16 += (long)HID * 64;
  _Float16* gcn_wh[3];
  for (int i = 0; i < 3; ++i) { gcn_wh[i] = p16; p16 += (long)HID * HID; }
  _Float16* gin_w1h[3], * gin_w2h[3];
  for (int i = 0; i < 3; ++i) {
    gin_w1h[i] = p16; p16 += (long)2 * HID * HID;
    gin_w2h[i] = p16; p16 += (long)HID * 2 * HID;
  }
  _Float16* qwh = p16; p16 += (long)HID * HID;
  _Float16* kwh = p16; p16 += (long)HID * HID;
  _Float16* vwh = p16; p16 += (long)HID * HID;
  _Float16* owh = p16; p16 += (long)HID * HID;

  // ---- one-time weight conversion ----
  cvt16(ae_w, ae_wh, (long)HID * 128, stream);
  cvt16(me_w, me_wh, (long)HID * 64,  stream);
  for (int i = 0; i < 3; ++i) cvt16(gcn_w[i], gcn_wh[i], (long)HID * HID, stream);
  for (int i = 0; i < 3; ++i) {
    cvt16(gin_w1[i], gin_w1h[i], (long)2 * HID * HID, stream);
    cvt16(gin_w2[i], gin_w2h[i], (long)HID * 2 * HID, stream);
  }
  cvt16(qw, qwh, (long)HID * HID, stream);
  cvt16(kw, kwh, (long)HID * HID, stream);
  cvt16(vw, vwh, (long)HID * HID, stream);
  cvt16(ow, owh, (long)HID * HID, stream);

  // encoders
  linear_h(atom_f,  ae_wh, ae_b, x, N_ATOMS,  HID, 128, 0, stream);
  linear_h(motif_f, me_wh, me_b, m, N_MOTIFS, HID, 64,  0, stream);

  // GCN (atoms)
  k_fill<<<cdiv(N_ATOMS, 256), 256, 0, stream>>>(deg, 1.0f, N_ATOMS);
  k_deg<<<cdiv(E_ATOM, 256), 256, 0, stream>>>(dst, deg, E_ATOM);
  k_rsqrt<<<cdiv(N_ATOMS, 256), 256, 0, stream>>>(deg, N_ATOMS);
  for (int l = 0; l < 3; ++l) {
    linear_h(x, gcn_wh[l], nullptr, t0, N_ATOMS, HID, HID, 0, stream);
    zero(t1, A, stream);
    long nthr = (long)E_ATOM * (HID / 4);
    k_gcn_scatter<<<cdiv(nthr, 256), 256, 0, stream>>>(t0, src, dst, deg, t1, E_ATOM, HID);
    k_gcn_finish<<<cdiv(A, 256), 256, 0, stream>>>(t1, t0, deg, gcn_b[l], x, N_ATOMS, HID);
  }

  // GIN (motifs)
  const long Am = (long)N_MOTIFS * HID;
  for (int l = 0; l < 3; ++l) {
    zero(mt1, Am, stream);
    long nthr = (long)E_MOTIF * (HID / 4);
    k_gin_scatter<<<cdiv(nthr, 256), 256, 0, stream>>>(m, msrc, mdst, mt1, E_MOTIF, HID);
    k_add<<<cdiv(Am, 256), 256, 0, stream>>>(m, mt1, mt2, Am);
    linear_h(mt2, gin_w1h[l], gin_b1[l], mt0, N_MOTIFS, 2 * HID, HID,     1, stream);
    linear_h(mt0, gin_w2h[l], gin_b2[l], m,   N_MOTIFS, HID,     2 * HID, 1, stream);
  }

  // MHA (atoms)
  linear_h(x, qwh, qb, t0, N_ATOMS, HID, HID, 0, stream);
  linear_h(x, kwh, kb, t1, N_ATOMS, HID, HID, 0, stream);
  linear_h(x, vwh, vb, t2, N_ATOMS, HID, HID, 0, stream);
  cvt16(t0, qh, A, stream);
  cvt16(t1, kh, A, stream);
  k_transpose_f16<<<cdiv(A, 256), 256, 0, stream>>>(t2, vT, N_ATOMS, HID);
  k_flash<<<dim3(N_ATOMS / 16, HEADS), 32, 0, stream>>>(qh, kh, vT, t0, N_ATOMS);
  linear_h(t0, owh, ob, t1, N_ATOMS, HID, HID, 0, stream);
  k_colmean<<<cdiv(HID, 256), 256, 0, stream>>>(t1, comb, N_ATOMS, HID);

  // MHA (motifs)
  linear_h(m, qwh, qb, t0, N_MOTIFS, HID, HID, 0, stream);
  linear_h(m, kwh, kb, t1, N_MOTIFS, HID, HID, 0, stream);
  linear_h(m, vwh, vb, t2, N_MOTIFS, HID, HID, 0, stream);
  cvt16(t0, qh, Am, stream);
  cvt16(t1, kh, Am, stream);
  k_transpose_f16<<<cdiv(Am, 256), 256, 0, stream>>>(t2, vT, N_MOTIFS, HID);
  k_flash<<<dim3(N_MOTIFS / 16, HEADS), 32, 0, stream>>>(qh, kh, vT, t0, N_MOTIFS);
  linear_h(t0, owh, ob, t1, N_MOTIFS, HID, HID, 0, stream);
  k_colmean<<<cdiv(HID, 256), 256, 0, stream>>>(t1, comb + HID, N_MOTIFS, HID);

  // projection head (M=1, f32 weights)
  k_linear_small<<<dim3(HID / 16, 1), 128, 0, stream>>>(comb, pw1, pb1, hprj, 1, HID, 2 * HID, 1);
  k_linear_small<<<dim3(LAT / 16, 1), 128, 0, stream>>>(hprj, pw2, pb2, (float*)d_out, 1, LAT, HID, 0);
}
